// QuantizedLinear_3822520893772
// MI455X (gfx1250) — matrix operand order repaired and checked by reading, not compiled, not run on hardware
//
#include <hip/hip_runtime.h>

// ---------------------------------------------------------------------------
// QuantizedLinear for MI455X (gfx1250, wave32, WMMA).
//   rt = bf16(Pi^T); Wq = bf16(C[idx]);
//   y  = x @ Pi;  out = y @ Wq^T + bias      (bf16 WMMA, f32 accum)
// GEMM: block tile 128(M)x256(N), BK=32, 8 waves, wave tile 64x64 (4x4 frags,
// 16 v_wmma_f32_16x16x32_bf16 per wave per K-step). B (and A in GEMM2) staged
// with GLOBAL_LOAD_ASYNC_TO_LDS_B128; fp32->bf16 via V_CVT_PK_BF16_F32.
// Workspace: y (64MB) + Wq (32MB) + rt (32MB) = 128MB.
// ---------------------------------------------------------------------------

typedef __attribute__((ext_vector_type(16))) __bf16          v16bf;
typedef __attribute__((ext_vector_type(2)))  __bf16          v2bf;
typedef __attribute__((ext_vector_type(8)))  float           v8f;
typedef __attribute__((ext_vector_type(4)))  int             v4i;
typedef __attribute__((ext_vector_type(8)))  unsigned short  u16x8;
typedef __attribute__((ext_vector_type(16))) unsigned short  u16x16;

#if __has_builtin(__builtin_amdgcn_global_load_async_to_lds_b128)
#define HAS_ASYNC_LDS 1
#else
#define HAS_ASYNC_LDS 0
#endif

// two fp32 -> packed bf16 pair (one dword)
static __device__ inline unsigned cvt_pk_bf16(float a, float b) {
#if __has_builtin(__builtin_amdgcn_cvt_pk_bf16_f32)
  v2bf r = __builtin_amdgcn_cvt_pk_bf16_f32(a, b);
  union { v2bf v; unsigned u; } c; c.v = r; return c.u;
#else
  union { v2bf v; unsigned u; } c;
  c.v[0] = (__bf16)a; c.v[1] = (__bf16)b; return c.u;
#endif
}

static __device__ inline unsigned short f2bf1(float f) {
  union { __bf16 b; unsigned short u; } c; c.b = (__bf16)f; return c.u;
}

static __device__ inline v16bf mk16(u16x8 lo, u16x8 hi) {
  u16x16 u = __builtin_shufflevector(lo, hi,
      0, 1, 2, 3, 4, 5, 6, 7, 8, 9, 10, 11, 12, 13, 14, 15);
  union { u16x16 u; v16bf b; } c; c.u = u; return c.b;
}

// 16-byte global -> LDS copy (async on CDNA5, ASYNCcnt-tracked)
static __device__ inline void async_b128(const unsigned short* g, unsigned short* l) {
#if HAS_ASYNC_LDS
  __builtin_amdgcn_global_load_async_to_lds_b128(
      (__attribute__((address_space(1))) v4i*)g,
      (__attribute__((address_space(3))) v4i*)l, 0, 0);
#else
  *(u16x8*)l = *(const u16x8*)g;
#endif
}

static __device__ inline void wait_async() {
#if HAS_ASYNC_LDS
#if __has_builtin(__builtin_amdgcn_s_wait_asynccnt)
  __builtin_amdgcn_s_wait_asynccnt(0);
#else
  asm volatile("s_wait_asynccnt 0x0" ::: "memory");
#endif
#endif
}

// ---------------------------------------------------------------------------
// rt[n*K + k] = bf16(src[k*N + n])   (64x64 tiles through LDS)
// ---------------------------------------------------------------------------
__global__ __launch_bounds__(256) void transpose_cvt_kernel(
    const float* __restrict__ src, unsigned short* __restrict__ dst,
    int N, int K) {
  __shared__ float tile[64][68];
  const int k0 = blockIdx.y * 64;
  const int n0 = blockIdx.x * 64;
  const int tr = threadIdx.x >> 4;         // 0..15
  const int tc = (threadIdx.x & 15) * 4;   // 0..60
#pragma unroll
  for (int i = 0; i < 4; ++i) {
    const int kr = tr + i * 16;
    float4 v = *(const float4*)(src + (size_t)(k0 + kr) * N + n0 + tc);
    tile[kr][tc + 0] = v.x; tile[kr][tc + 1] = v.y;
    tile[kr][tc + 2] = v.z; tile[kr][tc + 3] = v.w;
  }
  __syncthreads();
#pragma unroll
  for (int i = 0; i < 4; ++i) {
    const int nr = tr + i * 16;
    uint2 o = make_uint2(cvt_pk_bf16(tile[tc + 0][nr], tile[tc + 1][nr]),
                         cvt_pk_bf16(tile[tc + 2][nr], tile[tc + 3][nr]));
    *(uint2*)(dst + (size_t)(n0 + nr) * K + k0 + tc) = o;
  }
}

// ---------------------------------------------------------------------------
// Wq[o*K + d] = bf16(codebook[idx[o*K + d]])
// ---------------------------------------------------------------------------
__global__ __launch_bounds__(256) void dequant_kernel(
    const int* __restrict__ idx, const float* __restrict__ cb,
    unsigned short* __restrict__ wq, int n) {
  __shared__ float cbs[16];
  if (threadIdx.x < 16) cbs[threadIdx.x] = cb[threadIdx.x];
  __syncthreads();
  size_t base = ((size_t)blockIdx.x * 256 + threadIdx.x) * 8;
  if (base + 8 <= (size_t)n) {
    int4 i0 = *(const int4*)(idx + base);
    int4 i1 = *(const int4*)(idx + base + 4);
    uint4 o = make_uint4(
        cvt_pk_bf16(cbs[i0.x & 15], cbs[i0.y & 15]),
        cvt_pk_bf16(cbs[i0.z & 15], cbs[i0.w & 15]),
        cvt_pk_bf16(cbs[i1.x & 15], cbs[i1.y & 15]),
        cvt_pk_bf16(cbs[i1.z & 15], cbs[i1.w & 15]));
    *(uint4*)(wq + base) = o;
  }
}

// ---------------------------------------------------------------------------
// GEMM: C[m,n] = sum_k A[m,k]*B[n,k]  (B given row-major [N,K], bf16)
// FIRST=true : A = x fp32 [M,K] (register-staged + v_cvt_pk); C = y bf16.
// FIRST=false: A = y bf16 [M,K] (async-LDS staged);           C = out f32+bias.
// Block 128x256, BK=32, 256 threads = 8 waves (2Mx4N), wave tile 64x64.
// ---------------------------------------------------------------------------
template <bool FIRST>
__global__ __launch_bounds__(256, 1) void gemm_bf16_wmma(
    const void* __restrict__ Ap, const unsigned short* __restrict__ Bp,
    const float* __restrict__ bias, void* __restrict__ Cp,
    int M, int N, int K) {
  constexpr int BM = 128, BN = 256, BK = 32;
  constexpr int AST = 40, BST = 40;  // padded LDS row strides (elements)

  __shared__ __align__(16) unsigned short As[2][BM * AST];  // 2 x 10KB
  __shared__ __align__(16) unsigned short Bs[2][BN * BST];  // 2 x 20KB

  const int tid  = threadIdx.x;
  const int lane = tid & 31;
  const int wid  = tid >> 5;
  const int m0 = blockIdx.y * BM;
  const int n0 = blockIdx.x * BN;

  const int wm = (wid & 1) * 64;   // 2 M-groups
  const int wn = (wid >> 1) * 64;  // 4 N-groups
  const int lrow  = lane & 15;
  const int lhalf = lane >> 4;

  // A staging: thread -> row 0..127, 16-elem K chunk (0/16)
  const int srow = tid >> 1;
  const int skc  = (tid & 1) * 16;

  float fa[16];

  auto stageB = [&](int buf, int kt2) {   // one row per thread, first 16 of its 32 k, async
    const unsigned short* src = Bp + (size_t)(n0 + tid) * K + (size_t)kt2 * BK;
    unsigned short* dst = &Bs[buf][tid * BST];
    async_b128(src, dst);
    async_b128(src + 8, dst + 8);
  };
  auto stageA_async = [&](int buf, int kt2) {
    const unsigned short* A = (const unsigned short*)Ap;
    const unsigned short* src =
        A + (size_t)(m0 + srow) * K + (size_t)kt2 * BK + skc;
    unsigned short* dst = &As[buf][srow * AST + skc];
    async_b128(src, dst);
    async_b128(src + 8, dst + 8);
  };
  auto loadA_f32 = [&](int kt2) {
    const float* A = (const float*)Ap;
    const float4* p =
        (const float4*)(A + (size_t)(m0 + srow) * K + (size_t)kt2 * BK + skc);
#pragma unroll
    for (int i = 0; i < 4; ++i) *(float4*)&fa[4 * i] = p[i];
  };
  auto storeA_f32 = [&](int buf) {
    unsigned w[8];
#pragma unroll
    for (int i = 0; i < 8; ++i) w[i] = cvt_pk_bf16(fa[2 * i], fa[2 * i + 1]);
    *(uint4*)&As[buf][srow * AST + skc]     = make_uint4(w[0], w[1], w[2], w[3]);
    *(uint4*)&As[buf][srow * AST + skc + 8] = make_uint4(w[4], w[5], w[6], w[7]);
  };

  // prologue: stage tile 0
  if (FIRST) { loadA_f32(0); storeA_f32(0); } else { stageA_async(0, 0); }
  stageB(0, 0);
  wait_async();
  __syncthreads();

  v8f acc[4][4] = {};
  const int nk = K / BK;

  for (int kt = 0; kt < nk; ++kt) {
    const int cur = kt & 1;
    const bool more = (kt + 1 < nk);

    // issue next tile's data movement (overlaps WMMA below)
    if (more) {
      if (FIRST) loadA_f32(kt + 1); else stageA_async(cur ^ 1, kt + 1);
      stageB(cur ^ 1, kt + 1);
    }

    const unsigned short* as = &As[cur][0];
    const unsigned short* bs = &Bs[cur][0];

    v16bf a[4];
#pragma unroll
    for (int mf = 0; mf < 4; ++mf) {
      int base = (wm + mf * 16 + lrow) * AST + lhalf * 8;
      a[mf] = mk16(*(const u16x8*)(as + base), *(const u16x8*)(as + base + 16));
    }
    v16bf b[4];
#pragma unroll
    for (int nf = 0; nf < 4; ++nf) {
      int base = (wn + nf * 16 + lrow) * BST + lhalf * 8;
      b[nf] = mk16(*(const u16x8*)(bs + base), *(const u16x8*)(bs + base + 16));
    }

#pragma unroll
    for (int mf = 0; mf < 4; ++mf)
#pragma unroll
      for (int nf = 0; nf < 4; ++nf)
        acc[mf][nf] = __builtin_amdgcn_wmma_f32_16x16x32_bf16(
            false, a[mf], false, b[nf], (short)0, acc[mf][nf], false, false);

    if (more && FIRST) storeA_f32(cur ^ 1);
    wait_async();
    __syncthreads();
  }

  // Epilogue: VGPR r -> M = 8*half + r, N = lane%16
#pragma unroll
  for (int mf = 0; mf < 4; ++mf) {
#pragma unroll
    for (int nf = 0; nf < 4; ++nf) {
      const int col  = n0 + wn + nf * 16 + lrow;
      const int rowb = m0 + wm + mf * 16 + lhalf * 8;
      if (FIRST) {
        unsigned short* C = (unsigned short*)Cp;
#pragma unroll
        for (int r = 0; r < 8; ++r)
          C[(size_t)(rowb + r) * N + col] = f2bf1(acc[mf][nf][r]);
      } else {
        float* C = (float*)Cp;
        const float bv = bias[col];
#pragma unroll
        for (int r = 0; r < 8; ++r)
          C[(size_t)(rowb + r) * N + col] = acc[mf][nf][r] + bv;
      }
    }
  }
}

// ---------------------------------------------------------------------------
extern "C" void kernel_launch(void* const* d_in, const int* in_sizes, int n_in,
                              void* d_out, int out_size, void* d_ws, size_t ws_size,
                              hipStream_t stream) {
  (void)n_in; (void)out_size; (void)ws_size;
  const float* x    = (const float*)d_in[0];   // [B,S,IN_F] fp32
  const int*   idx  = (const int*)d_in[1];     // [OUT_F,IN_F] int32
  const float* cb   = (const float*)d_in[2];   // [16] fp32
  const float* rot  = (const float*)d_in[3];   // [IN_F,IN_F] fp32
  const float* bias = (const float*)d_in[4];   // [OUT_F] fp32
  float* out = (float*)d_out;                  // [B,S,OUT_F] fp32

  const int IN_F = 4096, OUT_F = 4096;
  const int Mtot = in_sizes[0] / IN_F;         // B*S = 8192

  unsigned short* y  = (unsigned short*)d_ws;                        // [M, K] bf16
  unsigned short* wq = y  + (size_t)Mtot * IN_F;                     // [O, K] bf16
  unsigned short* rt = wq + (size_t)OUT_F * IN_F;                    // [N, K] bf16

  // rt = bf16(Pi^T)
  transpose_cvt_kernel<<<dim3(IN_F / 64, IN_F / 64), 256, 0, stream>>>(
      rot, rt, IN_F, IN_F);

  // Wq = bf16(C[idx])
  const int nW = OUT_F * IN_F;
  dequant_kernel<<<nW / (256 * 8), 256, 0, stream>>>(idx, cb, wq, nW);

  // y = x @ Pi
  gemm_bf16_wmma<true><<<dim3(IN_F / 256, Mtot / 128), 256, 0, stream>>>(
      x, rt, nullptr, y, Mtot, IN_F, IN_F);

  // out = y @ Wq^T + bias
  gemm_bf16_wmma<false><<<dim3(OUT_F / 256, Mtot / 128), 256, 0, stream>>>(
      y, wq, bias, out, Mtot, OUT_F, IN_F);
}
